// TripCenterLoss_min_margin_67937792688366
// MI455X (gfx1250) — matrix operand order repaired and checked by reading, not compiled, not run on hardware
//
#include <hip/hip_runtime.h>

// ---------------------------------------------------------------------------
// TripCenterLoss (min-margin) for MI455X / gfx1250, wave32 + WMMA bf16.
//
// distmat = |x|^2 + |c|^2 - 2 x@c^T   (B=8192, C=4096, D=2048)
// loss = mean(relu(margin + w*dist_own - (1-w)*min_other))
//
// GEMM done with V_WMMA_F32_16X16X32_BF16 using a 2-term bf16 split of the
// fp32 inputs (hi*hi + hi*lo + lo*hi) for ~fp32 accuracy at bf16-WMMA rate.
// ---------------------------------------------------------------------------

typedef __bf16 bf16_t;
typedef __attribute__((ext_vector_type(16))) __bf16        v16bf;
typedef __attribute__((ext_vector_type(8)))  float         v8f;
typedef __attribute__((ext_vector_type(4)))  unsigned int  v4u;

#define B_ROWS 8192
#define C_ROWS 4096
#define D_DIM  2048

// Monotonic float <-> uint mapping so atomicMin(u32) == float min.
static __device__ __forceinline__ unsigned int f2key(float f) {
  unsigned int b = __float_as_uint(f);
  return (b & 0x80000000u) ? ~b : (b | 0x80000000u);
}
static __device__ __forceinline__ float key2f(unsigned int k) {
  unsigned int b = (k & 0x80000000u) ? (k & 0x7fffffffu) : ~k;
  return __uint_as_float(b);
}

union FragU { v4u u[2]; v16bf v; };

// A fragment 16x32 bf16 (M x K), row-major source [rows, D_DIM].
// lane 0-15: M=lane, K pairs base 0 ; lane 16-31: M=lane-16, K base +8.
// VGPR0-3: K = kb..kb+7 (16B contiguous), VGPR4-7: K = kb+16..kb+23.
static __device__ __forceinline__ v16bf load_frag_a(const bf16_t* __restrict__ base,
                                                    int row, int k0, int lane) {
  const int m  = lane & 15;
  const int kb = (lane & 16) ? 8 : 0;
  const bf16_t* p = base + (size_t)(row + m) * D_DIM + (k0 + kb);
  FragU r;
  r.u[0] = *(const v4u*)(p);
  r.u[1] = *(const v4u*)(p + 16);
  return r.v;
}

// B fragment 32x16 bf16 (K x N) where B = centers^T tile; centers row-major
// [C_ROWS, D_DIM], so column n of B = row (col0+n) of centers, K contiguous.
// lane 0-15: N=lane, K = 0..7,16..23 ; lane 16-31: N=lane-16, K = 8..15,24..31.
static __device__ __forceinline__ v16bf load_frag_b(const bf16_t* __restrict__ base,
                                                    int col0, int k0, int lane) {
  const int n  = lane & 15;
  const int kh = (lane & 16) ? 8 : 0;
  const bf16_t* p = base + (size_t)(col0 + n) * D_DIM + (k0 + kh);
  FragU r;
  r.u[0] = *(const v4u*)(p);
  r.u[1] = *(const v4u*)(p + 16);
  return r.v;
}

// ---------------------------------------------------------------------------
// Kernel 1: fp32 -> bf16 hi/lo split + row squared-norm (fp32 exact).
// One block per row, 256 threads.
// ---------------------------------------------------------------------------
__global__ __launch_bounds__(256) void prep_kernel(const float* __restrict__ src,
                                                   bf16_t* __restrict__ hi,
                                                   bf16_t* __restrict__ lo,
                                                   float* __restrict__ nrm) {
  const int row = blockIdx.x;
  const size_t base = (size_t)row * D_DIM;
  float s = 0.0f;
  for (int i = threadIdx.x; i < D_DIM; i += 256) {
    float f  = src[base + i];
    bf16_t h = (bf16_t)f;
    float fh = (float)h;
    bf16_t l = (bf16_t)(f - fh);
    hi[base + i] = h;
    lo[base + i] = l;
    s += f * f;
  }
  __shared__ float red[256];
  red[threadIdx.x] = s;
  __syncthreads();
  #pragma unroll
  for (int st = 128; st > 0; st >>= 1) {
    if (threadIdx.x < st) red[threadIdx.x] += red[threadIdx.x + st];
    __syncthreads();
  }
  if (threadIdx.x == 0) nrm[row] = red[0];
}

// ---------------------------------------------------------------------------
// Kernel 2: init per-row min keys to "+infinity-or-worse".
// ---------------------------------------------------------------------------
__global__ __launch_bounds__(256) void init_kernel(unsigned int* __restrict__ minkey) {
  int i = blockIdx.x * 256 + threadIdx.x;
  if (i < B_ROWS) minkey[i] = 0xFFFFFFFFu;
}

// ---------------------------------------------------------------------------
// Kernel 3: fused distance GEMM + per-row min / own-class extraction.
// Block = 128(M) x 128(N) tile, 256 threads = 8 wave32s in a 2(M) x 4(N)
// grid; each wave owns 64x32 = 4x2 WMMA 16x16 accumulators.
// ---------------------------------------------------------------------------
__global__ __launch_bounds__(256, 1) void fused_distmat_kernel(
    const bf16_t* __restrict__ xh, const bf16_t* __restrict__ xl,
    const bf16_t* __restrict__ ch, const bf16_t* __restrict__ cl,
    const float*  __restrict__ nx, const float*  __restrict__ nc,
    const int*    __restrict__ labels,
    unsigned int* __restrict__ minkey,
    float*        __restrict__ down) {
  const int wave = threadIdx.x >> 5;
  const int lane = threadIdx.x & 31;
  const int wm = wave >> 2;              // 0..1  (M direction)
  const int wn = wave & 3;               // 0..3  (N direction)
  const int m0 = blockIdx.y * 128 + wm * 64;
  const int c0 = blockIdx.x * 128 + wn * 32;

  v8f acc[4][2];
  #pragma unroll
  for (int mt = 0; mt < 4; ++mt)
    #pragma unroll
    for (int nt = 0; nt < 2; ++nt)
      acc[mt][nt] = (v8f){};

  for (int k0 = 0; k0 < D_DIM; k0 += 32) {
    v16bf ah[4], al[4], bh[2], bl[2];
    #pragma unroll
    for (int mt = 0; mt < 4; ++mt) {
      ah[mt] = load_frag_a(xh, m0 + mt * 16, k0, lane);
      al[mt] = load_frag_a(xl, m0 + mt * 16, k0, lane);
    }
    #pragma unroll
    for (int nt = 0; nt < 2; ++nt) {
      bh[nt] = load_frag_b(ch, c0 + nt * 16, k0, lane);
      bl[nt] = load_frag_b(cl, c0 + nt * 16, k0, lane);
    }
    // bf16x3 split: hi*hi + hi*lo + lo*hi  (lo*lo term negligible)
    #pragma unroll
    for (int mt = 0; mt < 4; ++mt) {
      #pragma unroll
      for (int nt = 0; nt < 2; ++nt) {
        acc[mt][nt] = __builtin_amdgcn_wmma_f32_16x16x32_bf16(
            false, ah[mt], false, bh[nt], (short)0, acc[mt][nt], false, false);
        acc[mt][nt] = __builtin_amdgcn_wmma_f32_16x16x32_bf16(
            false, ah[mt], false, bl[nt], (short)0, acc[mt][nt], false, false);
        acc[mt][nt] = __builtin_amdgcn_wmma_f32_16x16x32_bf16(
            false, al[mt], false, bh[nt], (short)0, acc[mt][nt], false, false);
      }
    }
  }

  // Epilogue: C/D fragment layout -> (m, c) coordinates:
  //   VGPR r, lanes 0-15:  M = r,     N = lane
  //   VGPR r, lanes 16-31: M = r + 8, N = lane - 16
  const int nlane = lane & 15;
  const int mhalf = (lane & 16) ? 8 : 0;
  const float INF = __builtin_inff();

  float ncv[2];
  #pragma unroll
  for (int nt = 0; nt < 2; ++nt) ncv[nt] = nc[c0 + nt * 16 + nlane];

  #pragma unroll
  for (int mt = 0; mt < 4; ++mt) {
    #pragma unroll
    for (int r = 0; r < 8; ++r) {
      const int m    = m0 + mt * 16 + r + mhalf;
      const float nxv = nx[m];
      const int   lm  = labels[m];
      float vmin = INF;
      #pragma unroll
      for (int nt = 0; nt < 2; ++nt) {
        const int c = c0 + nt * 16 + nlane;
        float v = nxv + ncv[nt] - 2.0f * acc[mt][nt][r];
        if (c == lm) { down[m] = v; v = INF; }  // own class: record, mask
        vmin = fminf(vmin, v);
      }
      // min across the 16 lanes of this half (xor masks stay within halves)
      #pragma unroll
      for (int s = 1; s < 16; s <<= 1)
        vmin = fminf(vmin, __shfl_xor(vmin, s, 32));
      if (nlane == 0) atomicMin(&minkey[m], f2key(vmin));
    }
  }
}

// ---------------------------------------------------------------------------
// Kernel 4: hinge + mean -> scalar. Single block.
// ---------------------------------------------------------------------------
__global__ __launch_bounds__(256) void finalize_kernel(
    const unsigned int* __restrict__ minkey,
    const float* __restrict__ down,
    const float* __restrict__ margin,
    const float* __restrict__ weight,
    float* __restrict__ out) {
  const float mg = margin[0];
  const float w  = weight[0];
  float s = 0.0f;
  for (int i = threadIdx.x; i < B_ROWS; i += 256) {
    float dmin = key2f(minkey[i]);
    float h = mg + w * down[i] - (1.0f - w) * dmin;
    s += fmaxf(h, 0.0f);
  }
  __shared__ float red[256];
  red[threadIdx.x] = s;
  __syncthreads();
  #pragma unroll
  for (int st = 128; st > 0; st >>= 1) {
    if (threadIdx.x < st) red[threadIdx.x] += red[threadIdx.x + st];
    __syncthreads();
  }
  if (threadIdx.x == 0) out[0] = red[0] / (float)B_ROWS;
}

// ---------------------------------------------------------------------------
extern "C" void kernel_launch(void* const* d_in, const int* in_sizes, int n_in,
                              void* d_out, int out_size, void* d_ws, size_t ws_size,
                              hipStream_t stream) {
  (void)in_sizes; (void)n_in; (void)out_size; (void)ws_size;

  const float* x       = (const float*)d_in[0];   // [B, D]
  const float* centers = (const float*)d_in[1];   // [C, D]
  const int*   labels  = (const int*)d_in[2];     // [B]
  const float* margin  = (const float*)d_in[3];   // [1]
  const float* weight  = (const float*)d_in[4];   // [1]
  float* out = (float*)d_out;

  // Workspace layout (~96.1 MB total)
  char* ws = (char*)d_ws;
  bf16_t* xh = (bf16_t*)ws;  ws += (size_t)B_ROWS * D_DIM * sizeof(bf16_t);
  bf16_t* xl = (bf16_t*)ws;  ws += (size_t)B_ROWS * D_DIM * sizeof(bf16_t);
  bf16_t* ch = (bf16_t*)ws;  ws += (size_t)C_ROWS * D_DIM * sizeof(bf16_t);
  bf16_t* cl = (bf16_t*)ws;  ws += (size_t)C_ROWS * D_DIM * sizeof(bf16_t);
  float*  nx = (float*)ws;   ws += (size_t)B_ROWS * sizeof(float);
  float*  nc = (float*)ws;   ws += (size_t)C_ROWS * sizeof(float);
  unsigned int* minkey = (unsigned int*)ws; ws += (size_t)B_ROWS * sizeof(unsigned int);
  float*  down = (float*)ws; ws += (size_t)B_ROWS * sizeof(float);

  prep_kernel<<<B_ROWS, 256, 0, stream>>>(x, xh, xl, nx);
  prep_kernel<<<C_ROWS, 256, 0, stream>>>(centers, ch, cl, nc);
  init_kernel<<<B_ROWS / 256, 256, 0, stream>>>(minkey);

  dim3 grid(C_ROWS / 128, B_ROWS / 128);  // (32, 64)
  fused_distmat_kernel<<<grid, 256, 0, stream>>>(xh, xl, ch, cl, nx, nc,
                                                 labels, minkey, down);

  finalize_kernel<<<1, 256, 0, stream>>>(minkey, down, margin, weight, out);
}
